// TConv_6949257085180
// MI455X (gfx1250) — 8-bit matrix operand order repaired and checked by reading, compiled, not run on hardware
//
#include <hip/hip_runtime.h>

typedef __attribute__((ext_vector_type(16))) int   v16i;
typedef __attribute__((ext_vector_type(8)))  float v8f;

#define CIN   32
#define COUT  16
#define HIN   128
#define WIN   128
#define HOUT  256
#define WOUT  256
#define ROW_DW 33
#define NROWS 6

__global__ __launch_bounds__(512) void bin_tconv_wmma(
    const float* __restrict__ x, const float* __restrict__ w, float* __restrict__ out)
{
  __shared__ __align__(16) unsigned int xbw[NROWS][CIN][ROW_DW];
  __shared__ __align__(64) unsigned int ldsA[2][2][32][16];
  __shared__ __align__(16) float        ldsD[2][COUT][WOUT];

  const int tid  = threadIdx.x;
  const int lane = tid & 31;
  const int wv   = tid >> 5;
  const int px   = wv >> 3;
  const int tx   = wv & 7;

  const unsigned bi = blockIdx.x;
  const int s  = bi & 31;
  const int n  = bi >> 5;
  const int oy0 = s << 3;
  const int iyb = 4 * s - 1;

  const float* xn = x + (size_t)n * (CIN * HIN * WIN);

  if (tid < CIN * NROWS) {
    const int ci = tid / NROWS;
    const int lr = tid - ci * NROWS;
    const int iy = 4 * (s + 1) - 1 + lr;
    if (iy >= 0 && iy < HIN)
      __builtin_prefetch(xn + (ci * HIN + iy) * WIN, 0, 1);
  }

  for (int d = tid; d < NROWS * CIN * ROW_DW; d += 512) {
    const int lr  = d / (CIN * ROW_DW);
    const int rem = d - lr * (CIN * ROW_DW);
    const int ci  = rem / ROW_DW;
    const int dw  = rem - ci * ROW_DW;
    const int iy  = iyb + lr;
    unsigned word = 0;
    if ((unsigned)iy < (unsigned)HIN) {
      const float* rowp = xn + (ci * HIN + iy) * WIN;
#pragma unroll
      for (int b = 0; b < 4; ++b) {
        const int ix = dw * 4 + b - 1;
        if ((unsigned)ix < (unsigned)WIN) {
          const float v = rowp[ix];
          word |= ((v > 0.0f) ? 0x38u : 0xB8u) << (8 * b);
        }
      }
    }
    xbw[lr][ci][dw] = word;
  }

#pragma unroll
  for (int t = 0; t < 4; ++t) {
    const int d   = tid * 4 + t;
    const int pya = d >> 10;
    const int pxa = (d >> 9) & 1;
    const int sl  = (d >> 4) & 31;
    const int v   = d & 15;
    const int co  = sl & 15;
    const int h   = sl >> 4;
    const int v7  = v & 7;
    const int kb  = 64 * (v >> 3) + (v7 & 1) * 4 + (v7 >> 1) * 16 + 8 * h;
    const int ky0 = pya ? 0 : 1, ky1 = pya ? 2 : 3;
    const int kx0 = pxa ? 0 : 1, kx1 = pxa ? 2 : 3;
    unsigned word = 0;
#pragma unroll
    for (int b = 0; b < 4; ++b) {
      const int k  = kb + b;
      const int ci = k >> 2;
      const int ky = ((k >> 1) & 1) ? ky1 : ky0;
      const int kx = (k & 1) ? kx1 : kx0;
      const float wval = w[((ci * COUT + co) * 4 + ky) * 4 + kx];
      word |= ((wval > 0.0f) ? 0x38u : 0xB8u) << (8 * b);
    }
    ldsA[pya][pxa][sl][v] = word;
  }

  __syncthreads();

  const v16i A_e = *(const v16i*)(&ldsA[0][px][lane][0]);
  const v16i A_o = *(const v16i*)(&ldsA[1][px][lane][0]);

  const int m  = lane & 15;
  const int h  = lane >> 4;
  const int p0 = tx * 16 + m + px + 1;
  const unsigned char* xbb = (const unsigned char*)xbw;

  const int ox = 2 * (tx * 16 + m) + px;
  float* outn = out + ((size_t)n * COUT) * (HOUT * WOUT);

  auto assembleB = [&](int t) -> v16i {
    const unsigned char* b0 = xbb + (size_t)(t + 1) * (CIN * ROW_DW * 4);
    const unsigned char* b1 = xbb + (size_t)t * (CIN * ROW_DW * 4);
    v16i B;
#pragma unroll
    for (int v = 0; v < 16; ++v) {
      const int ci   = (v >> 3) * 16 + ((v >> 1) & 3) * 4 + 2 * h + (v & 1);
      const int base = ci * (ROW_DW * 4) + p0;
      unsigned word;
      word  = (unsigned)b0[base];
      word |= (unsigned)b0[base - 1] << 8;
      word |= (unsigned)b1[base] << 16;
      word |= (unsigned)b1[base - 1] << 24;
      B[v] = (int)word;
    }
    return B;
  };

  const v8f C0 = {0.f, 0.f, 0.f, 0.f, 0.f, 0.f, 0.f, 0.f};

  v16i Bprev = assembleB(0);
#pragma unroll
  for (int u = 0; u < 4; ++u) {
    const v16i Bnext = assembleB(u + 1);

    const v8f D0 = __builtin_amdgcn_wmma_f32_16x16x128_fp8_fp8(
        A_e, Bprev, (short)0, C0, false, false);
    const v8f D1 = __builtin_amdgcn_wmma_f32_16x16x128_fp8_fp8(
        A_o, Bnext, (short)0, C0, false, false);

#pragma unroll
    for (int q = 0; q < 8; ++q) {
      ldsD[0][q + 8 * h][ox] = D0[q];
      ldsD[1][q + 8 * h][ox] = D1[q];
    }

    __syncthreads();

    {
      const int rr = tid >> 8;
      const int co = (tid >> 4) & 15;
      const int o  = (tid & 15) * 16;
      const int oy = oy0 + 2 * u + rr;
      const float4* src = (const float4*)&ldsD[rr][co][o];
      float4* dst = (float4*)(outn + (size_t)co * (HOUT * WOUT) + (size_t)oy * WOUT + o);
#pragma unroll
      for (int j = 0; j < 4; ++j) dst[j] = src[j];
    }

    __syncthreads();
    Bprev = Bnext;
  }
}

extern "C" void kernel_launch(void* const* d_in, const int* in_sizes, int n_in,
                              void* d_out, int out_size, void* d_ws, size_t ws_size,
                              hipStream_t stream) {
  const float* x = (const float*)d_in[0];
  const float* w = (const float*)d_in[1];
  float* out = (float*)d_out;
  (void)in_sizes; (void)n_in; (void)out_size; (void)d_ws; (void)ws_size;

  const int blocks = 32 * 32;
  bin_tconv_wmma<<<blocks, 512, 0, stream>>>(x, w, out);
}
